// Fai_score_7653631721538
// MI455X (gfx1250) — matrix operand order repaired and checked by reading, not compiled, not run on hardware
//
#include <hip/hip_runtime.h>
#include <hip/hip_bf16.h>

typedef __attribute__((ext_vector_type(16))) _Float16 v16h;
typedef __attribute__((ext_vector_type(4)))  _Float16 v4h;
typedef __attribute__((ext_vector_type(8)))  float    v8f;

#define EMBD 300
#define NFILT 128
#define LDSH  (18 * EMBD + 40)   // staged rows + slack for contiguous A reads

// ---------------------------------------------------------------------------
// Pad + convert conv weights: [128, Kreal] f32 -> [128, Kpad] f16 (zero pad)
// ---------------------------------------------------------------------------
__global__ void fai_pad_weights(const float* __restrict__ W, _Float16* __restrict__ Wp,
                                int Kreal, int Kpad) {
    int idx = blockIdx.x * blockDim.x + threadIdx.x;
    int total = NFILT * Kpad;
    if (idx >= total) return;
    int f = idx / Kpad;
    int j = idx - f * Kpad;
    Wp[idx] = (j < Kreal) ? (_Float16)W[f * Kreal + j] : (_Float16)0.0f;
}

// ---------------------------------------------------------------------------
// Embed -> im2col conv (WMMA GEMM) -> bias -> ReLU -> sum over positions.
// One block = one (sequence, 16-position tile). 4 waves, each owns 2 N-tiles.
// im2col row for position p0+laneM is CONTIGUOUS in LDS at xs + laneM*300,
// so A fragments are plain ds_load_b64 runs; K-pad handled by zero-fill +
// zero-padded B, so the inner loop has no guards at all.
// ---------------------------------------------------------------------------
template <int KPAD, int KW>
__global__ __launch_bounds__(128) void fai_conv_wmma(
    const int*      __restrict__ tokens,  // [B, L]
    const float*    __restrict__ embed,   // [VOCAB, 300]
    const _Float16* __restrict__ Wp,      // [128, KPAD] zero-padded f16
    const float*    __restrict__ bias,    // [128]
    float*          __restrict__ acc,     // [B, 128]
    int L)
{
    const int P      = L - KW + 1;
    const int nTiles = (P + 15) >> 4;
    const int b      = blockIdx.x / nTiles;
    const int tile   = blockIdx.x - b * nTiles;
    const int p0     = tile << 4;
    const int rowsNeeded = min(L - p0, 15 + KW);

    __shared__ _Float16 xs[LDSH];

    // stage embedding rows (gather + f32->f16), zero-fill the tail
    const int total = rowsNeeded * EMBD;
    for (int idx = threadIdx.x; idx < total; idx += blockDim.x) {
        int r = idx / EMBD;
        int e = idx - r * EMBD;
        int tok = tokens[b * L + p0 + r];
        xs[idx] = (_Float16)embed[tok * EMBD + e];
    }
    for (int idx = total + threadIdx.x; idx < LDSH; idx += blockDim.x)
        xs[idx] = (_Float16)0.0f;
    __syncthreads();

    const int lane   = threadIdx.x & 31;
    const int wave   = threadIdx.x >> 5;
    const int laneM  = lane & 15;
    const int hiHalf = lane >> 4;
    const int col0   = (wave * 2) * 16 + laneM;   // waves cover N-tiles {0,1}..{6,7}
    const int col1   = col0 + 16;

    const _Float16* arow = xs + laneM * EMBD + hiHalf * 16;           // 8B aligned
    const _Float16* brow0 = Wp + (size_t)col0 * KPAD + hiHalf * 16;   // 32B aligned
    const _Float16* brow1 = Wp + (size_t)col1 * KPAD + hiHalf * 16;

    v8f c0 = {}; v8f c1 = {};
    constexpr int NCH = KPAD / 32;
#pragma unroll 4
    for (int kc = 0; kc < NCH; ++kc) {
        // --- A fragment: one contiguous 16-half run (K = base+{0..15}) ---
        const _Float16* ap = arow + kc * 32;
        v4h l0 = *(const v4h*)(ap);
        v4h l1 = *(const v4h*)(ap + 4);
        v4h h0 = *(const v4h*)(ap + 8);
        v4h h1 = *(const v4h*)(ap + 12);
        v16h a;
#pragma unroll
        for (int t = 0; t < 4; ++t) {
            a[t]      = l0[t];
            a[4 + t]  = l1[t];
            a[8 + t]  = h0[t];
            a[12 + t] = h1[t];
        }
        // --- B fragments: 16 contiguous K halves of column colX ---
        v16h b0 = *(const v16h*)(brow0 + kc * 32);
        v16h b1 = *(const v16h*)(brow1 + kc * 32);

        c0 = __builtin_amdgcn_wmma_f32_16x16x32_f16(false, a, false, b0, (short)0, c0, false, false);
        c1 = __builtin_amdgcn_wmma_f32_16x16x32_f16(false, a, false, b1, (short)0, c1, false, false);
    }

    // bias + ReLU + masked position-sum, then accumulate into per-seq mean acc
    float s0 = 0.f, s1 = 0.f;
    const float bb0 = bias[col0], bb1 = bias[col1];
#pragma unroll
    for (int r = 0; r < 8; ++r) {
        int pos = p0 + hiHalf * 8 + r;            // C layout: M = hiHalf*8 + vgpr
        if (pos < P) {
            s0 += fmaxf(c0[r] + bb0, 0.f);
            s1 += fmaxf(c1[r] + bb1, 0.f);
        }
    }
    atomicAdd(&acc[b * NFILT + col0], s0);
    atomicAdd(&acc[b * NFILT + col1], s1);
}

// ---------------------------------------------------------------------------
// acc/P -> L2-normalize each row (cosine prep): out = v / max(||row||, 1e-6)
// ---------------------------------------------------------------------------
__global__ __launch_bounds__(128) void fai_mean_norm(
    const float* __restrict__ acc, float* __restrict__ outn, float invP)
{
    const int row = blockIdx.x;
    const int f = threadIdx.x;
    float v = acc[row * NFILT + f] * invP;
    __shared__ float red[NFILT];
    __shared__ float nrm;
    red[f] = v * v;
    __syncthreads();
    for (int s = 64; s > 0; s >>= 1) {
        if (f < s) red[f] += red[f + s];
        __syncthreads();
    }
    if (f == 0) nrm = fmaxf(sqrtf(red[0]), 1e-6f);
    __syncthreads();
    outn[row * NFILT + f] = v / nrm;
}

// ---------------------------------------------------------------------------
// cosine matrices for mention rows: st, ct, sb, cb  [64 x 128]
// ---------------------------------------------------------------------------
__global__ __launch_bounds__(128) void fai_cos_pairs(
    const float* __restrict__ msn, const float* __restrict__ mcn,
    const float* __restrict__ etn, const float* __restrict__ ebn,
    float* __restrict__ st, float* __restrict__ ct,
    float* __restrict__ sb, float* __restrict__ cb)
{
    const int i = blockIdx.x;    // mention 0..63
    const int j = threadIdx.x;   // entity  0..127
    __shared__ float sms[NFILT], smc[NFILT];
    sms[j] = msn[i * NFILT + j];
    smc[j] = mcn[i * NFILT + j];
    __syncthreads();
    float a = 0.f, bv = 0.f, c = 0.f, d = 0.f;
    for (int t = 0; t < NFILT; ++t) {
        float et = etn[j * NFILT + t], eb = ebn[j * NFILT + t];
        a += sms[t] * et;  bv += smc[t] * et;
        c += sms[t] * eb;  d  += smc[t] * eb;
    }
    const int o = i * NFILT + j;
    st[o] = a; ct[o] = bv; sb[o] = c; cb[o] = d;
}

// doc-row cosines (broadcast over mentions): dt[j], db[j]
__global__ __launch_bounds__(128) void fai_cos_doc(
    const float* __restrict__ mdn, const float* __restrict__ etn,
    const float* __restrict__ ebn, float* __restrict__ dt, float* __restrict__ db)
{
    const int j = threadIdx.x;
    float a = 0.f, bv = 0.f;
    for (int t = 0; t < NFILT; ++t) {
        float md = mdn[t];
        a += md * etn[j * NFILT + t];
        bv += md * ebn[j * NFILT + t];
    }
    dt[j] = a; db[j] = bv;
}

// ---------------------------------------------------------------------------
// combine: x[i,j] = sfeat[i,j,:]·Wc[:623] + cos terms·Wc[623:629] + b
// one wave per (i,j) pair
// ---------------------------------------------------------------------------
__global__ __launch_bounds__(256) void fai_combine(
    const float* __restrict__ sfeat, const float* __restrict__ Wc,
    const float* __restrict__ bcomb,
    const float* __restrict__ st, const float* __restrict__ dtv,
    const float* __restrict__ ct, const float* __restrict__ sb,
    const float* __restrict__ dbv, const float* __restrict__ cb,
    float* __restrict__ x)
{
    const int wid  = blockIdx.x * 8 + (threadIdx.x >> 5);   // pair id 0..8191
    const int lane = threadIdx.x & 31;
    if (wid >= 64 * 128) return;
    const float* sf = sfeat + (size_t)wid * 623;
    float s = 0.f;
    for (int t = lane; t < 623; t += 32) s += sf[t] * Wc[t];
#pragma unroll
    for (int off = 16; off > 0; off >>= 1) s += __shfl_down(s, off, 32);
    if (lane == 0) {
        const int j = wid & 127;
        s += st[wid] * Wc[623] + dtv[j] * Wc[624] + ct[wid] * Wc[625]
           + sb[wid] * Wc[626] + dbv[j] * Wc[627] + cb[wid] * Wc[628] + bcomb[0];
        x[wid] = s;
    }
}

// ---------------------------------------------------------------------------
// per-row stats: z (ddof=1), softmax, min-max uniform, sm_s (row0 double-softmax)
// out = [z | sm | u | sm_s], each 64*128
// ---------------------------------------------------------------------------
__global__ __launch_bounds__(128) void fai_rowstats(
    const float* __restrict__ x, float* __restrict__ out)
{
    const int i = blockIdx.x;   // 0..63
    const int j = threadIdx.x;  // 0..127
    __shared__ float red[NFILT];
    const float v = x[i * NFILT + j];

    auto rsum = [&](float val) -> float {
        red[j] = val; __syncthreads();
        for (int s = 64; s > 0; s >>= 1) { if (j < s) red[j] += red[j + s]; __syncthreads(); }
        float r = red[0]; __syncthreads(); return r;
    };
    auto rmax = [&](float val) -> float {
        red[j] = val; __syncthreads();
        for (int s = 64; s > 0; s >>= 1) { if (j < s) red[j] = fmaxf(red[j], red[j + s]); __syncthreads(); }
        float r = red[0]; __syncthreads(); return r;
    };
    auto rmin = [&](float val) -> float {
        red[j] = val; __syncthreads();
        for (int s = 64; s > 0; s >>= 1) { if (j < s) red[j] = fminf(red[j], red[j + s]); __syncthreads(); }
        float r = red[0]; __syncthreads(); return r;
    };

    const float mean = rsum(v) * (1.0f / 128.0f);
    const float var  = rsum((v - mean) * (v - mean)) * (1.0f / 127.0f);  // ddof=1
    const float z    = (v - mean) / sqrtf(var);

    const float zmax = rmax(z);
    const float e    = __expf(z - zmax);
    const float sm   = e / rsum(e);

    const float xmin = rmin(v);
    const float xmax = rmax(v);
    float uu = (v + 1.0f - xmin) / (xmax - xmin);
    const float u = uu / rsum(uu);

    float sms = sm;
    if (i == 0) {                      // block-uniform: barriers OK
        const float m2 = rmax(sm);
        const float e2 = __expf(sm - m2);
        sms = e2 / rsum(e2);
    }
    const int o = i * NFILT + j;
    out[o]             = z;
    out[8192 + o]      = sm;
    out[16384 + o]     = u;
    out[24576 + o]     = sms;
}

// ---------------------------------------------------------------------------
extern "C" void kernel_launch(void* const* d_in, const int* in_sizes, int n_in,
                              void* d_out, int out_size, void* d_ws, size_t ws_size,
                              hipStream_t stream) {
    (void)in_sizes; (void)n_in; (void)out_size; (void)ws_size;
    const int*   mention_tok = (const int*)d_in[0];    // [64,10]
    const int*   context_tok = (const int*)d_in[1];    // [64,100]
    const int*   doc_tok     = (const int*)d_in[2];    // [512]
    const int*   title_tok   = (const int*)d_in[3];    // [128,20]
    const int*   body_tok    = (const int*)d_in[4];    // [128,512]
    const float* sfeatures   = (const float*)d_in[5];  // [64,128,623]
    const float* embed_table = (const float*)d_in[6];  // [5053,300]
    const float* W_ms = (const float*)d_in[7];   const float* b_ms = (const float*)d_in[8];
    const float* W_mc = (const float*)d_in[9];   const float* b_mc = (const float*)d_in[10];
    const float* W_md = (const float*)d_in[11];  const float* b_md = (const float*)d_in[12];
    const float* W_et = (const float*)d_in[13];  const float* b_et = (const float*)d_in[14];
    const float* W_eb = (const float*)d_in[15];  const float* b_eb = (const float*)d_in[16];
    const float* W_comb = (const float*)d_in[17]; const float* b_comb = (const float*)d_in[18];
    float* out = (float*)d_out;

    constexpr int KP2 = 608;   // k=2: 600 -> 608
    constexpr int KP3 = 928;   // k=3: 900 -> 928

    // workspace layout (256B-aligned slices)
    char* ws = (char*)d_ws;
    size_t off = 0;
    auto take = [&](size_t bytes) -> void* {
        void* p = ws + off;
        off = (off + bytes + 255) & ~(size_t)255;
        return p;
    };
    _Float16* Wp_ms = (_Float16*)take((size_t)NFILT * KP2 * 2);
    _Float16* Wp_mc = (_Float16*)take((size_t)NFILT * KP3 * 2);
    _Float16* Wp_md = (_Float16*)take((size_t)NFILT * KP3 * 2);
    _Float16* Wp_et = (_Float16*)take((size_t)NFILT * KP3 * 2);
    _Float16* Wp_eb = (_Float16*)take((size_t)NFILT * KP3 * 2);

    float* acc_all = (float*)take((size_t)(64 + 64 + 1 + 128 + 128) * NFILT * 4);
    float* acc_ms = acc_all;              // 64 rows
    float* acc_mc = acc_ms + 64  * NFILT;
    float* acc_md = acc_mc + 64  * NFILT; // 1 row
    float* acc_et = acc_md + 1   * NFILT;
    float* acc_eb = acc_et + 128 * NFILT;

    float* nrm_all = (float*)take((size_t)(64 + 64 + 1 + 128 + 128) * NFILT * 4);
    float* msn = nrm_all;
    float* mcn = msn + 64  * NFILT;
    float* mdn = mcn + 64  * NFILT;
    float* etn = mdn + 1   * NFILT;
    float* ebn = etn + 128 * NFILT;

    float* st = (float*)take(64 * 128 * 4);
    float* ct = (float*)take(64 * 128 * 4);
    float* sb = (float*)take(64 * 128 * 4);
    float* cb = (float*)take(64 * 128 * 4);
    float* dt = (float*)take(128 * 4);
    float* db = (float*)take(128 * 4);
    float* xv = (float*)take(64 * 128 * 4);

    // 1) weight pad/convert (f32 -> f16, K padded to mult of 32)
    {
        int t2 = NFILT * KP2, t3 = NFILT * KP3;
        fai_pad_weights<<<(t2 + 255) / 256, 256, 0, stream>>>(W_ms, Wp_ms, 600, KP2);
        fai_pad_weights<<<(t3 + 255) / 256, 256, 0, stream>>>(W_mc, Wp_mc, 900, KP3);
        fai_pad_weights<<<(t3 + 255) / 256, 256, 0, stream>>>(W_md, Wp_md, 900, KP3);
        fai_pad_weights<<<(t3 + 255) / 256, 256, 0, stream>>>(W_et, Wp_et, 900, KP3);
        fai_pad_weights<<<(t3 + 255) / 256, 256, 0, stream>>>(W_eb, Wp_eb, 900, KP3);
    }

    // 2) zero the position-sum accumulators (fresh every call; capture-safe)
    hipMemsetAsync(acc_all, 0, (size_t)(64 + 64 + 1 + 128 + 128) * NFILT * 4, stream);

    // 3) conv branches (WMMA). grid = B * ceil(P/16)
    fai_conv_wmma<KP2, 2><<<64  * 1,  128, 0, stream>>>(mention_tok, embed_table, Wp_ms, b_ms, acc_ms,  10); // P=9
    fai_conv_wmma<KP3, 3><<<64  * 7,  128, 0, stream>>>(context_tok, embed_table, Wp_mc, b_mc, acc_mc, 100); // P=98
    fai_conv_wmma<KP3, 3><<<1   * 32, 128, 0, stream>>>(doc_tok,     embed_table, Wp_md, b_md, acc_md, 512); // P=510
    fai_conv_wmma<KP3, 3><<<128 * 2,  128, 0, stream>>>(title_tok,   embed_table, Wp_et, b_et, acc_et,  20); // P=18
    fai_conv_wmma<KP3, 3><<<128 * 32, 128, 0, stream>>>(body_tok,    embed_table, Wp_eb, b_eb, acc_eb, 512); // P=510

    // 4) mean over positions + L2 row-normalize
    fai_mean_norm<<<64,  128, 0, stream>>>(acc_ms, msn, 1.0f / 9.0f);
    fai_mean_norm<<<64,  128, 0, stream>>>(acc_mc, mcn, 1.0f / 98.0f);
    fai_mean_norm<<<1,   128, 0, stream>>>(acc_md, mdn, 1.0f / 510.0f);
    fai_mean_norm<<<128, 128, 0, stream>>>(acc_et, etn, 1.0f / 18.0f);
    fai_mean_norm<<<128, 128, 0, stream>>>(acc_eb, ebn, 1.0f / 510.0f);

    // 5) cosine matrices
    fai_cos_pairs<<<64, 128, 0, stream>>>(msn, mcn, etn, ebn, st, ct, sb, cb);
    fai_cos_doc<<<1, 128, 0, stream>>>(mdn, etn, ebn, dt, db);

    // 6) combine (one wave per pair; 8192 pairs)
    fai_combine<<<1024, 256, 0, stream>>>(sfeatures, W_comb, b_comb,
                                          st, dt, ct, sb, db, cb, xv);

    // 7) per-row normalizations -> d_out = [z | sm | u | sm_s]
    fai_rowstats<<<64, 128, 0, stream>>>(xv, out);
}
